// TFFMambaBlock_84722524881205
// MI455X (gfx1250) — compile-verified, matrix operand order checked by reading, not run on hardware
//
#include <hip/hip_runtime.h>
#include <hip/hip_bf16.h>

// ---------------------------------------------------------------------------
// TFFMambaBlock forward for gfx1250 (MI455X).
// GEMMs: F16 operands (pre-packed, zero-padded) -> v_wmma_f32_16x16x32_f16,
// F32 accumulation; each wave computes NT adjacent 16x16 tiles reusing its
// A fragment. Scan/conv/gating in F32 VALU with async global->LDS staging.
// ---------------------------------------------------------------------------

typedef __attribute__((ext_vector_type(16))) _Float16 v16h;
typedef __attribute__((ext_vector_type(8)))  _Float16 v8h;
typedef __attribute__((ext_vector_type(8)))  float    v8f;

#define T_LEN 512
#define F_LEN 128
#define NTOK  (T_LEN * F_LEN)      // 65536 tokens per branch
#define SCHUNK 16                  // scan steps staged per LDS chunk

#if defined(__has_builtin)
#if __has_builtin(__builtin_amdgcn_global_load_async_to_lds_b32) && \
    __has_builtin(__builtin_amdgcn_s_wait_asynccnt)
#define HAVE_ASYNC_LDS 1
#endif
#endif

typedef __attribute__((address_space(1))) int g_int;
typedef __attribute__((address_space(3))) int l_int;

__device__ __forceinline__ float silu_f(float x)     { return x / (1.f + __expf(-x)); }
__device__ __forceinline__ float softplus_f(float x) { return x > 20.f ? x : log1pf(__expf(x)); }

// ---------------------------------------------------------------------------
// WMMA GEMM:  C[M, N] = act( A[M,K] * W^T + bias ) (+ res)
// A: f16 (M,lda); W: f16 (N,ldw) pre-packed+padded; N%16==0, K%32==0.
// Block (32,4): 4 waves; each wave owns NT adjacent 16x16 C tiles along N,
// reusing one A fragment per K-chunk (2 A-loads + 2*NT W-loads + NT wmma).
// A frag (ISA 7.12.2, 16-bit A 16x32): lane M = lane&15, K-half by lane>>4;
// B mirrors A. All dims padded -> straight b128 loads, no guards.
// ---------------------------------------------------------------------------
template<int ACT, bool BIAS, bool RES, bool OUT16, int NT>
__global__ __launch_bounds__(128)
void wmma_gemm(const _Float16* __restrict__ A, const _Float16* __restrict__ W,
               const float* __restrict__ bias, const float* __restrict__ res,
               float* __restrict__ Cmat, _Float16* __restrict__ C16,
               int K, int lda, int ldw, int ldc, int ldc16)
{
    const int lane  = threadIdx.x;                    // 0..31 (wave32)
    const int mtile = blockIdx.x * 4 + threadIdx.y;
    const int ntile = blockIdx.y * NT;
    const int r16   = lane & 15;
    const int hi    = lane >> 4;
    const int khalf = hi * 8;

    const int row = mtile * 16 + r16;

    const _Float16* __restrict__ Ap = A + (size_t)row * lda + khalf;
    const _Float16* Wp[NT];
#pragma unroll
    for (int u = 0; u < NT; ++u)
        Wp[u] = W + (size_t)((ntile + u) * 16 + r16) * ldw + khalf;

    v8f acc[NT];
#pragma unroll
    for (int u = 0; u < NT; ++u) acc[u] = (v8f){};

    for (int k0 = 0; k0 < K; k0 += 32) {
        __builtin_prefetch(Ap + k0 + 64, 0, 3);
        const v8h a0 = *(const v8h*)(Ap + k0);        // K = k0+khalf+0..7
        const v8h a1 = *(const v8h*)(Ap + k0 + 16);   // K = k0+16+khalf+0..7
        const v16h va = __builtin_shufflevector(a0, a1, 0,1,2,3,4,5,6,7,8,9,10,11,12,13,14,15);
#pragma unroll
        for (int u = 0; u < NT; ++u) {
            const v8h b0 = *(const v8h*)(Wp[u] + k0);
            const v8h b1 = *(const v8h*)(Wp[u] + k0 + 16);
            const v16h vb = __builtin_shufflevector(b0, b1, 0,1,2,3,4,5,6,7,8,9,10,11,12,13,14,15);
            acc[u] = __builtin_amdgcn_wmma_f32_16x16x32_f16(
                         false, va, false, vb, (short)0, acc[u], false, false);
        }
    }

#pragma unroll
    for (int u = 0; u < NT; ++u) {
        const int col = (ntile + u) * 16 + r16;
        const float bv = BIAS ? bias[col] : 0.f;
#pragma unroll
        for (int r = 0; r < 8; ++r) {
            const int mOut = mtile * 16 + r + hi * 8; // C layout: VGPR r -> M=r / r+8
            float v = acc[u][r] + bv;
            if (ACT == 1) v = softplus_f(v);
            if (RES) v += res[(size_t)mOut * ldc + col];
            Cmat[(size_t)mOut * ldc + col] = v;
            if (OUT16) C16[(size_t)mOut * ldc16 + col] = (_Float16)v;
        }
    }
}

// ---------------------------------------------------------------------------
// Weight pack: f32 (Nsrc,Ksrc) [or its transpose] -> f16 (Ndst,Kdst), 0-padded
// ---------------------------------------------------------------------------
__global__ __launch_bounds__(256)
void pack_w16(const float* __restrict__ W, _Float16* __restrict__ Wp,
              int Ndst, int Kdst, int Nsrc, int Ksrc, int trans)
{
    int i = blockIdx.x * 256 + threadIdx.x;
    if (i >= Ndst * Kdst) return;
    int n = i / Kdst, k = i - n * Kdst;
    float v = 0.f;
    if (n < Nsrc && k < Ksrc)
        v = trans ? W[(size_t)k * Nsrc + n] : W[(size_t)n * Ksrc + k];
    Wp[i] = (_Float16)v;
}

// ---------------------------------------------------------------------------
// Layout shuffles (dual f32 + f16 outputs where the result feeds a GEMM)
// ---------------------------------------------------------------------------
__global__ __launch_bounds__(256)
void pack_time(const float* __restrict__ x, float* __restrict__ xin,
               _Float16* __restrict__ xin16)
{
    int i = blockIdx.x * 256 + threadIdx.x;           // over NTOK*32
    int c  = i & 31;
    int ft = i >> 5;
    int f  = ft / T_LEN;
    int t  = ft - f * T_LEN;
    float v = x[((size_t)c * T_LEN + t) * F_LEN + f];
    xin[i] = v; xin16[i] = (_Float16)v;
}

__global__ __launch_bounds__(256)
void pack_freq(const float* __restrict__ xf1, float* __restrict__ xin,
               _Float16* __restrict__ xin16)
{
    int i = blockIdx.x * 256 + threadIdx.x;
    int c  = i & 31;
    int tf = i >> 5;
    int t  = tf / F_LEN;
    int f  = tf - t * F_LEN;
    float v = xf1[((size_t)f * T_LEN + t) * 32 + c];
    xin[i] = v; xin16[i] = (_Float16)v;
}

__global__ __launch_bounds__(256)
void unpack_out(const float* __restrict__ xf2, float* __restrict__ out)
{
    int i = blockIdx.x * 256 + threadIdx.x;           // over C*T*F
    int c   = i / (T_LEN * F_LEN);
    int rem = i - c * (T_LEN * F_LEN);
    int t   = rem / F_LEN;
    int f   = rem - t * F_LEN;
    out[i] = xf2[((size_t)t * F_LEN + f) * 32 + c];
}

// ---------------------------------------------------------------------------
// Depthwise causal conv1d(k=4) + bias + SiLU over xm = xz[:,0:256] (ld 512)
// ---------------------------------------------------------------------------
__global__ __launch_bounds__(256)
void dwconv_silu(const float* __restrict__ xz, const float* __restrict__ w,
                 const float* __restrict__ b, float* __restrict__ xc,
                 _Float16* __restrict__ xc16, int L)
{
    int i   = blockIdx.x * 256 + threadIdx.x;         // over NTOK*256
    int d   = i & 255;
    int tok = i >> 8;
    int t   = tok % L;
    float s = b[d];
#pragma unroll
    for (int j = 0; j < 4; ++j) {
        int tt = t - 3 + j;
        if (tt >= 0) s += w[d * 4 + j] * xz[(size_t)(tok - 3 + j) * 512 + d];
    }
    float v = silu_f(s);
    xc[i] = v; xc16[i] = (_Float16)v;
}

// ---------------------------------------------------------------------------
// Selective scan, one sequence per block (256 threads = d_inner), state in
// VGPRs. B/C staged through LDS in SCHUNK-step chunks; async copy to LDS
// (ASYNCcnt-tracked) when the toolchain exposes the gfx1250 builtins.
// xdbl row stride 64: cols 4..19 = B, 20..35 = C.
// ---------------------------------------------------------------------------
__global__ __launch_bounds__(256)
void ssm_scan(const float* __restrict__ xc, const float* __restrict__ dt,
              const float* __restrict__ xdbl, const float* __restrict__ A_log,
              float* __restrict__ ys, int L)
{
    const int seq = blockIdx.x;
    const int d   = threadIdx.x;
    __shared__ float sBC[SCHUNK * 32];

    float A[16], h[16];
#pragma unroll
    for (int s = 0; s < 16; ++s) { A[s] = -__expf(A_log[d * 16 + s]); h[s] = 0.f; }

    for (int t0 = 0; t0 < L; t0 += SCHUNK) {
        __syncthreads();                              // prev chunk consumed
        if (d < 32) {                                 // wave 0 stages B/C
            const float* gsrc = xdbl + ((size_t)seq * L + t0) * 64 + 4 + d;
#ifdef HAVE_ASYNC_LDS
#pragma unroll
            for (int s = 0; s < SCHUNK; ++s)
                __builtin_amdgcn_global_load_async_to_lds_b32(
                    (g_int*)(gsrc + (size_t)s * 64),
                    (l_int*)&sBC[s * 32 + d],
                    0, 0);
            __builtin_amdgcn_s_wait_asynccnt(0);
#else
#pragma unroll
            for (int s = 0; s < SCHUNK; ++s)
                sBC[s * 32 + d] = gsrc[(size_t)s * 64];
#endif
        }
        __syncthreads();

#pragma unroll 1
        for (int s = 0; s < SCHUNK; ++s) {
            const size_t tok = (size_t)seq * L + t0 + s;
            const float u     = xc[tok * 256 + d];
            const float delta = dt[tok * 256 + d];
            const float du    = delta * u;
            float acc = 0.f;
#pragma unroll
            for (int st = 0; st < 16; ++st) {
                h[st] = h[st] * __expf(delta * A[st]) + du * sBC[s * 32 + st];
                acc += h[st] * sBC[s * 32 + 16 + st];
            }
            ys[tok * 256 + d] = acc;
        }
    }
}

// y = (ys + xc*D) * silu(z), z = xz[:,256+d]; in-place f32 + f16 copy
__global__ __launch_bounds__(256)
void gate_silu(float* __restrict__ ys, _Float16* __restrict__ ys16,
               const float* __restrict__ xc, const float* __restrict__ Dp,
               const float* __restrict__ xz)
{
    int i   = blockIdx.x * 256 + threadIdx.x;         // over NTOK*256
    int d   = i & 255;
    int tok = i >> 8;
    float z = xz[(size_t)tok * 512 + 256 + d];
    float v = (ys[i] + xc[i] * Dp[d]) * silu_f(z);
    ys[i] = v; ys16[i] = (_Float16)v;
}

// m = gamma * tanh(alpha*m + beta1[ch]) + beta[ch]; in-place f32 + f16 copy
__global__ __launch_bounds__(256)
void dytanh(float* __restrict__ m, _Float16* __restrict__ m16,
            const float* __restrict__ alpha, const float* __restrict__ beta,
            const float* __restrict__ gamma, const float* __restrict__ beta1)
{
    int i  = blockIdx.x * 256 + threadIdx.x;          // over NTOK*64
    int ch = i & 63;
    float v = gamma[0] * tanhf(alpha[0] * m[i] + beta1[ch]) + beta[ch];
    m[i] = v; m16[i] = (_Float16)v;
}

// ---------------------------------------------------------------------------
// Host-side orchestration
// ---------------------------------------------------------------------------
struct BranchP {
    const float *conv_w, *conv_b;
    const float *in_proj_w, *m_conv_w, *m_conv_b, *x_proj_w,
                *dt_proj_w, *dt_proj_b, *A_log, *D, *out_proj_w;
    const float *alpha, *beta, *gamma, *beta1, *lin_w, *lin_b;
};

static BranchP load_branch(void* const* d_in, int base)
{
    BranchP p;
    p.conv_w     = (const float*)d_in[base + 0];
    p.conv_b     = (const float*)d_in[base + 1];
    p.in_proj_w  = (const float*)d_in[base + 2];
    p.m_conv_w   = (const float*)d_in[base + 3];
    p.m_conv_b   = (const float*)d_in[base + 4];
    p.x_proj_w   = (const float*)d_in[base + 5];
    p.dt_proj_w  = (const float*)d_in[base + 6];
    p.dt_proj_b  = (const float*)d_in[base + 7];
    p.A_log      = (const float*)d_in[base + 8];
    p.D          = (const float*)d_in[base + 9];
    p.out_proj_w = (const float*)d_in[base + 10];
    p.alpha      = (const float*)d_in[base + 11];
    p.beta       = (const float*)d_in[base + 12];
    p.gamma      = (const float*)d_in[base + 13];
    p.beta1      = (const float*)d_in[base + 14];
    p.lin_w      = (const float*)d_in[base + 15];
    p.lin_b      = (const float*)d_in[base + 16];
    return p;
}

struct Ws {
    // f32
    float *y64, *xz, *xc, *dtb, *xdbl, *ys, *m64, *xin;
    // f16 activations
    _Float16 *xin16, *y64h, *xc16, *xdbl16, *ys16, *m16;
    // f16 packed weights (reused per branch)
    _Float16 *wc1, *win, *wxp, *wdt, *wout, *wlin;
};

static void run_branch(const BranchP& p, const Ws& w, float* out, int L,
                       hipStream_t stream)
{
    const dim3 gblk(32, 4);
    const int  GM = NTOK / 64;

    // ---- pack weights into padded f16 (tiny) ----
    pack_w16<<< 8, 256, 0, stream>>>(p.conv_w,     w.wc1,  64,  32, 64, 32, 0);
    pack_w16<<<128, 256, 0, stream>>>(p.in_proj_w, w.win, 512,  64, 512, 64, 0);
    pack_w16<<< 48, 256, 0, stream>>>(p.x_proj_w,  w.wxp,  48, 256, 36, 256, 0);
    pack_w16<<< 32, 256, 0, stream>>>(p.dt_proj_w, w.wdt, 256,  32, 256, 4, 0);
    pack_w16<<< 64, 256, 0, stream>>>(p.out_proj_w,w.wout, 64, 256, 64, 256, 0);
    pack_w16<<<  8, 256, 0, stream>>>(p.lin_w,     w.wlin, 32,  64, 32, 64, 1);  // (K,N)->(N,K)

    // 1) y64 = xin @ conv_w^T + conv_b            (NTOK,64), K=32, NT=4
    wmma_gemm<0, true,  false, true,  4><<<dim3(GM, 1), gblk, 0, stream>>>(
        w.xin16, w.wc1, p.conv_b, nullptr, w.y64, w.y64h, 32, 32, 32, 64, 64);
    // 2) xz = y64 @ in_proj_w^T                   (NTOK,512), K=64, NT=4
    wmma_gemm<0, false, false, false, 4><<<dim3(GM, 8), gblk, 0, stream>>>(
        w.y64h, w.win, nullptr, nullptr, w.xz, nullptr, 64, 64, 64, 512, 0);
    // 3) depthwise causal conv + SiLU -> xc (f32 + f16)
    dwconv_silu<<<NTOK, 256, 0, stream>>>(w.xz, p.m_conv_w, p.m_conv_b, w.xc, w.xc16, L);
    // 4) xdbl = xc @ x_proj_w^T                   (NTOK,48 padded, ld64), K=256, NT=3
    wmma_gemm<0, false, false, true,  3><<<dim3(GM, 1), gblk, 0, stream>>>(
        w.xc16, w.wxp, nullptr, nullptr, w.xdbl, w.xdbl16, 256, 256, 256, 64, 64);
    // 5) dt = softplus(xdbl[:, :4] @ dt_proj_w^T + dt_proj_b)  (NTOK,256), K=32 pad, NT=4
    wmma_gemm<1, true,  false, false, 4><<<dim3(GM, 4), gblk, 0, stream>>>(
        w.xdbl16, w.wdt, p.dt_proj_b, nullptr, w.dtb, nullptr, 32, 64, 32, 256, 0);
    // 6) selective scan -> ys
    ssm_scan<<<NTOK / L, 256, 0, stream>>>(w.xc, w.dtb, w.xdbl, p.A_log, w.ys, L);
    // 7) ys = (ys + xc*D) * silu(z)  (f32 in-place + f16)
    gate_silu<<<NTOK, 256, 0, stream>>>(w.ys, w.ys16, w.xc, p.D, w.xz);
    // 8) m64 = ys @ out_proj_w^T + y64            (NTOK,64), K=256, NT=4
    wmma_gemm<0, false, true,  false, 4><<<dim3(GM, 1), gblk, 0, stream>>>(
        w.ys16, w.wout, nullptr, w.y64, w.m64, nullptr, 256, 256, 256, 64, 0);
    // 9) DyTanh in-place (f32 + f16)
    dytanh<<<NTOK * 64 / 256, 256, 0, stream>>>(w.m64, w.m16, p.alpha, p.beta, p.gamma, p.beta1);
    // 10) out = m64 @ lin_w + lin_b + xin         (NTOK,32), K=64, NT=2
    wmma_gemm<0, true,  true,  false, 2><<<dim3(GM, 1), gblk, 0, stream>>>(
        w.m16, w.wlin, p.lin_b, w.xin, out, nullptr, 64, 64, 64, 32, 0);
}

extern "C" void kernel_launch(void* const* d_in, const int* in_sizes, int n_in,
                              void* d_out, int out_size, void* d_ws, size_t ws_size,
                              hipStream_t stream)
{
    (void)in_sizes; (void)n_in; (void)out_size; (void)ws_size;

    const float* x = (const float*)d_in[0];
    BranchP pt = load_branch(d_in, 1);        // params_t (17 leaves)
    BranchP pf = load_branch(d_in, 18);       // params_f (17 leaves)

    float* out = (float*)d_out;               // (1,32,512,128)
    float* xf1 = out + (size_t)NTOK * 32;     // x_f1 (128,512,32)
    float* xf2 = xf1 + (size_t)NTOK * 32;     // x_f2 (512,128,32)

    // ---- workspace carve-up ----
    float* ws = (float*)d_ws;
    Ws w; size_t o = 0;
    w.y64  = ws + o; o += (size_t)NTOK * 64;
    w.xz   = ws + o; o += (size_t)NTOK * 512;
    w.xc   = ws + o; o += (size_t)NTOK * 256;
    w.dtb  = ws + o; o += (size_t)NTOK * 256;
    w.xdbl = ws + o; o += (size_t)NTOK * 64;
    w.ys   = ws + o; o += (size_t)NTOK * 256;
    w.m64  = ws + o; o += (size_t)NTOK * 64;
    w.xin  = ws + o; o += (size_t)NTOK * 32;
    _Float16* h16 = (_Float16*)(ws + o); size_t q = 0;
    w.xin16  = h16 + q; q += (size_t)NTOK * 32;
    w.y64h   = h16 + q; q += (size_t)NTOK * 64;
    w.xc16   = h16 + q; q += (size_t)NTOK * 256;
    w.xdbl16 = h16 + q; q += (size_t)NTOK * 64;
    w.ys16   = h16 + q; q += (size_t)NTOK * 256;
    w.m16    = h16 + q; q += (size_t)NTOK * 64;
    w.wc1    = h16 + q; q += 64 * 32;
    w.win    = h16 + q; q += 512 * 64;
    w.wxp    = h16 + q; q += 48 * 256;
    w.wdt    = h16 + q; q += 256 * 32;
    w.wout   = h16 + q; q += 64 * 256;
    w.wlin   = h16 + q; q += 32 * 64;

    // ---- Branch 1: time-axis (128 sequences, L = 512) ----
    pack_time<<<NTOK * 32 / 256, 256, 0, stream>>>(x, w.xin, w.xin16);
    run_branch(pt, w, xf1, T_LEN, stream);

    // ---- Branch 2: freq-axis (512 sequences, L = 128) ----
    pack_freq<<<NTOK * 32 / 256, 256, 0, stream>>>(xf1, w.xin, w.xin16);
    run_branch(pf, w, xf2, F_LEN, stream);

    // ---- Final layout back to (1,C,T,F) ----
    unpack_out<<<NTOK * 32 / 256, 256, 0, stream>>>(xf2, out);
}
